// Net_MP_68805376082308
// MI455X (gfx1250) — compile-verified, matrix operand order checked by reading, not run on hardware
//
#include <hip/hip_runtime.h>

#define WIDTH 64
#define NN    10000
#define NE    50000
#define DEPTH 4

typedef __attribute__((ext_vector_type(16))) __bf16 v16bf;
typedef __attribute__((ext_vector_type(8)))  __bf16 v8bf;
typedef __attribute__((ext_vector_type(8)))  float  v8f;
typedef __attribute__((ext_vector_type(4)))  float  v4f;
typedef __attribute__((ext_vector_type(2)))  float  v2f;

// ---------------------------------------------------------------------------
// Small prep kernels
// ---------------------------------------------------------------------------

// h0[n,o] = x[n,:] @ fc1_W + fc1_b   (N x 3 @ 3 x 64)
__global__ void k_h0(const float* __restrict__ x, const float* __restrict__ W,
                     const float* __restrict__ b, float* __restrict__ h) {
    int idx = blockIdx.x * blockDim.x + threadIdx.x;
    if (idx >= NN * WIDTH) return;
    int n = idx >> 6, o = idx & 63;
    float acc = b[o];
    acc += x[n * 3 + 0] * W[0 * WIDTH + o];
    acc += x[n * 3 + 1] * W[1 * WIDTH + o];
    acc += x[n * 3 + 2] * W[2 * WIDTH + o];
    h[idx] = acc;
}

// z[e,o] = relu(edge_attr[e,:] @ k1_W + k1_b) in bf16   (E x 3 @ 3 x 64)
__global__ void k_z(const float* __restrict__ ea, const float* __restrict__ W,
                    const float* __restrict__ b, __bf16* __restrict__ z) {
    int idx = blockIdx.x * blockDim.x + threadIdx.x;
    if (idx >= NE * WIDTH) return;
    int e = idx >> 6, o = idx & 63;
    float acc = b[o];
    acc += ea[e * 3 + 0] * W[0 * WIDTH + o];
    acc += ea[e * 3 + 1] * W[1 * WIDTH + o];
    acc += ea[e * 3 + 2] * W[2 * WIDTH + o];
    z[idx] = (__bf16)(acc > 0.f ? acc : 0.f);
}

// k2T[j,c] = bf16(k2_W[c,j])  -> [4096, 64], so B-tile K-runs are contiguous
__global__ void k_k2t(const float* __restrict__ k2W, __bf16* __restrict__ k2t) {
    int idx = blockIdx.x * blockDim.x + threadIdx.x;
    if (idx >= WIDTH * WIDTH * WIDTH) return;   // 64*4096
    int j = idx >> 6, c = idx & 63;
    k2t[idx] = (__bf16)k2W[c * (WIDTH * WIDTH) + j];
}

__global__ void k_zero(float* __restrict__ p, int n) {
    int idx = blockIdx.x * blockDim.x + threadIdx.x;
    if (idx < n) p[idx] = 0.f;
}

__global__ void k_count(const int* __restrict__ dst, float* __restrict__ counts) {
    int e = blockIdx.x * blockDim.x + threadIdx.x;
    if (e < NE) atomicAdd(&counts[dst[e]], 1.0f);
}

// ---------------------------------------------------------------------------
// ew = z @ k2 + b2  via v_wmma_f32_16x16x32_bf16, stored bf16 [E, 4096].
// block = 256 threads (8 waves) handles 16 edges. 32 passes; per pass the 8
// waves compute 8 adjacent 16x16 tiles (128 cols), stage them in LDS, then
// all 256 threads write the 16x128 slab back with coalesced b128 stores
// (256 contiguous bytes per edge row).
// ---------------------------------------------------------------------------
__global__ void k_ew(const __bf16* __restrict__ z, const __bf16* __restrict__ k2t,
                     const float* __restrict__ b2, __bf16* __restrict__ ew) {
    __shared__ __align__(16) __bf16 st[16][128];   // 4 KB staging slab

    const int et   = blockIdx.x;          // edge tile (16 edges), 3125 tiles
    const int wave = threadIdx.x >> 5;    // 0..7
    const int lane = threadIdx.x & 31;
    const int l16  = lane & 15;
    const int hf   = lane >> 4;           // half-wave select

    // --- A operand: z rows, loaded once per wave -------------------------
    // chunk q covers K=[32q,32q+32); lane holds runs 32q+8*hf and 32q+16+8*hf
    const __bf16* zrow = z + (size_t)(et * 16 + l16) * WIDTH;
    v16bf a0, a1;
    {
        v8bf r0 = *(const v8bf*)(zrow + 8 * hf);
        v8bf r1 = *(const v8bf*)(zrow + 16 + 8 * hf);
        v8bf r2 = *(const v8bf*)(zrow + 32 + 8 * hf);
        v8bf r3 = *(const v8bf*)(zrow + 48 + 8 * hf);
        #pragma unroll
        for (int i = 0; i < 8; ++i) {
            a0[i] = r0[i]; a0[i + 8] = r1[i];
            a1[i] = r2[i]; a1[i + 8] = r3[i];
        }
    }

    const int srow = threadIdx.x >> 4;    // writeback mapping: edge row 0..15
    const int sc8  = threadIdx.x & 15;    // 8-col chunk 0..15

    for (int pass = 0; pass < 32; ++pass) {
        const int nt   = pass * 8 + wave;               // 16-col tile index
        const int colg = nt * 16 + l16;                 // global output column
        const __bf16* brow = k2t + (size_t)colg * WIDTH; // k2T[colg, 0..63]

        v16bf b0, b1;
        {
            v8bf r0 = *(const v8bf*)(brow + 8 * hf);
            v8bf r1 = *(const v8bf*)(brow + 16 + 8 * hf);
            v8bf r2 = *(const v8bf*)(brow + 32 + 8 * hf);
            v8bf r3 = *(const v8bf*)(brow + 48 + 8 * hf);
            #pragma unroll
            for (int i = 0; i < 8; ++i) {
                b0[i] = r0[i]; b0[i + 8] = r1[i];
                b1[i] = r2[i]; b1[i + 8] = r3[i];
            }
        }

        v8f c = {};
        c = __builtin_amdgcn_wmma_f32_16x16x32_bf16(false, a0, false, b0,
                                                    (short)0, c, false, false);
        c = __builtin_amdgcn_wmma_f32_16x16x32_bf16(false, a1, false, b1,
                                                    (short)0, c, false, false);

        const float bias = b2[colg];
        #pragma unroll
        for (int r = 0; r < 8; ++r)                     // C: M = r + 8*hf
            st[r + 8 * hf][wave * 16 + l16] = (__bf16)(c[r] + bias);

        __syncthreads();
        // coalesced writeback: 16 consecutive threads cover one edge row
        // (128 cols = 256 B), each storing one b128.
        *(v8bf*)(ew + (size_t)(et * 16 + srow) * 4096 + pass * 128 + sc8 * 8) =
            *(const v8bf*)&st[srow][sc8 * 8];
        __syncthreads();
    }
}

// ---------------------------------------------------------------------------
// msg[e,o] = sum_i h[src[e],i] * ew[e,i,o] ; scatter-add into agg[dst[e],o].
// Pure bandwidth on ew (8 KB/edge bf16). 8 threads per edge, each owning 8
// consecutive columns: per row i the 8 threads issue one b128 load each
// (128 B coalesced), 64 rows, no cross-thread reduction.
// 256-thread block handles 32 edges.
// ---------------------------------------------------------------------------
__global__ void k_msg(const float* __restrict__ h, const __bf16* __restrict__ ew,
                      const int* __restrict__ src, const int* __restrict__ dst,
                      float* __restrict__ agg) {
    __shared__ float hs[32][WIDTH];       // 8 KB gathered h[src]
    const int slot = threadIdx.x >> 3;    // edge slot in block, 0..31
    const int c8   = threadIdx.x & 7;     // 8-col chunk, 0..7
    const int e    = blockIdx.x * 32 + slot;

    if (e < NE) {
        const float* hrow = h + (size_t)src[e] * WIDTH;
        // each thread loads its 32B chunk of the row (2 x float4)
        *(v4f*)&hs[slot][c8 * 8]     = *(const v4f*)(hrow + c8 * 8);
        *(v4f*)&hs[slot][c8 * 8 + 4] = *(const v4f*)(hrow + c8 * 8 + 4);
    }
    __syncthreads();
    if (e >= NE) return;

    const __bf16* we = ew + (size_t)e * (WIDTH * WIDTH) + c8 * 8;
    float acc[8] = {0.f, 0.f, 0.f, 0.f, 0.f, 0.f, 0.f, 0.f};
    #pragma unroll 4
    for (int i = 0; i < WIDTH; ++i) {
        const v8bf w = *(const v8bf*)(we + i * WIDTH);
        const float hi = hs[slot][i];
        #pragma unroll
        for (int j = 0; j < 8; ++j) acc[j] += hi * (float)w[j];
    }

    float* ap = agg + (size_t)dst[e] * WIDTH + c8 * 8;
    #pragma unroll
    for (int j = 0; j < 8; ++j) atomicAdd(&ap[j], acc[j]);
}

// ---------------------------------------------------------------------------
// h_out = relu(agg/denom + h_in @ root + conv_b)
// root GEMM via v_wmma_f32_16x16x4_f32 (full f32). 625 node tiles, block of
// 128 threads = 4 waves, wave w owns output-column tile w (16 cols).
// ---------------------------------------------------------------------------
__global__ void k_update(const float* __restrict__ hin, const float* __restrict__ root,
                         const float* __restrict__ agg, const float* __restrict__ counts,
                         const float* __restrict__ convb, float* __restrict__ hout) {
    const int mt   = blockIdx.x;          // node tile (16 nodes)
    const int nt   = threadIdx.x >> 5;    // output tile 0..3
    const int lane = threadIdx.x & 31;
    const int l16  = lane & 15;
    const int hf   = lane >> 4;

    const int arow = mt * 16 + l16;       // A: M = l16
    const int colg = nt * 16 + l16;       // B: N = l16

    v8f acc = {};
    #pragma unroll
    for (int kk = 0; kk < 16; ++kk) {
        const int k0 = kk * 4 + 2 * hf;   // A/B VGPR0/1 hold K = k0, k0+1
        v2f a, b;
        a.x = hin[(size_t)arow * WIDTH + k0];
        a.y = hin[(size_t)arow * WIDTH + k0 + 1];
        b.x = root[(size_t)k0 * WIDTH + colg];
        b.y = root[(size_t)(k0 + 1) * WIDTH + colg];
        acc = __builtin_amdgcn_wmma_f32_16x16x4_f32(false, a, false, b,
                                                    (short)0, acc, false, false);
    }

    const float cb = convb[colg];
    #pragma unroll
    for (int r = 0; r < 8; ++r) {
        const int n = mt * 16 + r + 8 * hf;           // C: M = r + 8*hf
        const float cnt = counts[n];
        const float inv = cnt > 0.f ? 1.0f / cnt : 1.0f;
        float v = acc[r] + agg[(size_t)n * WIDTH + colg] * inv + cb;
        hout[(size_t)n * WIDTH + colg] = v > 0.f ? v : 0.f;
    }
}

// out[n] = h[n,:] @ fc2_W + fc2_b    ([N,64] @ [64,1])
__global__ void k_out(const float* __restrict__ h, const float* __restrict__ W,
                      const float* __restrict__ b, float* __restrict__ out) {
    int n = blockIdx.x * blockDim.x + threadIdx.x;
    if (n >= NN) return;
    float acc = b[0];
    #pragma unroll 8
    for (int i = 0; i < WIDTH; ++i) acc += h[(size_t)n * WIDTH + i] * W[i];
    out[n] = acc;
}

// ---------------------------------------------------------------------------

extern "C" void kernel_launch(void* const* d_in, const int* in_sizes, int n_in,
                              void* d_out, int out_size, void* d_ws, size_t ws_size,
                              hipStream_t stream) {
    const float* x     = (const float*)d_in[0];
    const int*   ei    = (const int*)  d_in[1];
    const float* ea    = (const float*)d_in[2];
    const float* fc1W  = (const float*)d_in[3];
    const float* fc1b  = (const float*)d_in[4];
    const float* k1W   = (const float*)d_in[5];
    const float* k1b   = (const float*)d_in[6];
    const float* k2W   = (const float*)d_in[7];
    const float* k2b   = (const float*)d_in[8];
    const float* root  = (const float*)d_in[9];
    const float* convb = (const float*)d_in[10];
    const float* fc2W  = (const float*)d_in[11];
    const float* fc2b  = (const float*)d_in[12];
    float* out = (float*)d_out;

    const int* src = ei;
    const int* dst = ei + NE;

    // workspace carving (256B aligned)
    char*  base = (char*)d_ws;
    size_t off  = 0;
    auto carve = [&](size_t bytes) -> char* {
        char* p = base + off;
        off += (bytes + 255) & ~(size_t)255;
        return p;
    };
    __bf16* k2t    = (__bf16*)carve((size_t)4096 * WIDTH * 2);        // 512 KB
    __bf16* z      = (__bf16*)carve((size_t)NE * WIDTH * 2);          // 6.4 MB
    __bf16* ew     = (__bf16*)carve((size_t)NE * 4096 * 2);           // 409.6 MB
    float*  hA     = (float*) carve((size_t)NN * WIDTH * 4);          // 2.56 MB
    float*  hB     = (float*) carve((size_t)NN * WIDTH * 4);          // 2.56 MB
    float*  agg    = (float*) carve((size_t)NN * WIDTH * 4);          // 2.56 MB
    float*  counts = (float*) carve((size_t)NN * 4);                  // 40 KB

    // --- prep -----------------------------------------------------------
    k_k2t<<<(4096 * WIDTH + 255) / 256, 256, 0, stream>>>(k2W, k2t);
    k_z  <<<(NE * WIDTH + 255) / 256, 256, 0, stream>>>(ea, k1W, k1b, z);
    k_h0 <<<(NN * WIDTH + 255) / 256, 256, 0, stream>>>(x, fc1W, fc1b, hA);
    k_zero<<<(NN + 255) / 256, 256, 0, stream>>>(counts, NN);
    k_count<<<(NE + 255) / 256, 256, 0, stream>>>(dst, counts);

    // --- ew GEMM (WMMA bf16) -------------------------------------------
    k_ew<<<NE / 16, 256, 0, stream>>>(z, k2t, k2b, ew);

    // --- DEPTH iterations of message passing ---------------------------
    float* hin  = hA;
    float* hout = hB;
    for (int d = 0; d < DEPTH; ++d) {
        k_zero<<<(NN * WIDTH + 255) / 256, 256, 0, stream>>>(agg, NN * WIDTH);
        k_msg<<<(NE + 31) / 32, 256, 0, stream>>>(hin, ew, src, dst, agg);
        k_update<<<NN / 16, 128, 0, stream>>>(hin, root, agg, counts, convb, hout);
        float* t = hin; hin = hout; hout = t;
    }

    // --- readout --------------------------------------------------------
    k_out<<<(NN + 255) / 256, 256, 0, stream>>>(hin, fc2W, fc2b, out);
}
